// NeuSDF_1743756722497
// MI455X (gfx1250) — matrix operand order repaired and checked by reading, not compiled, not run on hardware
//
#include <hip/hip_runtime.h>
#include <hip/hip_bf16.h>

#define RES 512
#define HID 128
#define PTS_PER_BLOCK 128
#define THREADS 256

typedef __attribute__((ext_vector_type(16))) _Float16 v16h;
typedef __attribute__((ext_vector_type(8)))  float    v8f;
typedef __attribute__((ext_vector_type(4)))  float    f32x4;
typedef __attribute__((ext_vector_type(2)))  float    f32x2;

union V16H {
    v16h  v;
    f32x4 q[2];
};

// ---------------------------------------------------------------------------
// Pre-pass: convert w2 (fp32, [K=128][N=128] row-major, h@w2 semantics) into
// fp16, pre-swizzled into the CDNA5 WMMA B-operand (32x16 f16) lane layout:
//   half index e = (((tile*4 + chunk)*32 + lane)*16 + i)
//   K   = chunk*32 + ((lane & 16) >> 1) + (i & 7) + ((i & 8) << 1)   (lanes 0-15: K 0..7,16..23; 16-31: K 8..15,24..31)
//   col = tile*16 + (lane & 15)
// Main kernel then fetches each B operand as 32 contiguous bytes per lane.
// ---------------------------------------------------------------------------
__global__ void swizzle_w2_kernel(const float* __restrict__ w2,
                                  _Float16* __restrict__ wsH) {
    int tid = threadIdx.x;                 // 1 block x 256 threads
    int base = tid * 64;                   // 16384 halves total
    for (int e = base; e < base + 64; ++e) {
        int i     = e & 15;
        int lane  = (e >> 4) & 31;
        int chunk = (e >> 9) & 3;
        int tile  = e >> 11;
        int K   = chunk * 32 + ((lane & 16) >> 1) + (i & 7) + ((i & 8) << 1);
        int col = tile * 16 + (lane & 15);
        wsH[e] = (_Float16)w2[K * HID + col];
    }
}

__device__ __forceinline__ float bilerp(const float* __restrict__ plane,
                                        float c1, float c2) {
    float f1 = floorf(c1), f2 = floorf(c2);
    int i1 = (int)f1, i2 = (int)f2;
    int c1c = min(i1 + 1, RES - 1);
    int c2c = min(i2 + 1, RES - 1);
    float bl = plane[i1  * RES + i2];
    float br = plane[c1c * RES + i2];
    float tl = plane[i1  * RES + c2c];
    float tr = plane[c1c * RES + c2c];
    float h = c1 - f1, v = c2 - f2;
    float top    = tl + (tr - tl) * h;
    float bottom = bl + (br - bl) * h;
    return bottom + (top - bottom) * v;
}

// ---------------------------------------------------------------------------
// Fused: tri-plane interp -> layer1 (fp32->fp16 LDS) -> WMMA layer2 ->
// fused bias/ReLU/layer3-dot in the C-tile register layout -> tanh -> out.
// 256 threads = 8 waves; 128 points per block; each wave owns 16 points.
// ---------------------------------------------------------------------------
__global__ __launch_bounds__(THREADS) void neusdf_fused_kernel(
    const float* __restrict__ pts,
    const float* __restrict__ xy, const float* __restrict__ yz,
    const float* __restrict__ xz,
    const float* __restrict__ w1, const float* __restrict__ b1,
    const float* __restrict__ b2, const float* __restrict__ w3,
    const float* __restrict__ b3,
    const _Float16* __restrict__ w2h,
    float* __restrict__ out, int nPoints)
{
    __shared__ float    sF[PTS_PER_BLOCK];
    __shared__ _Float16 sh1[PTS_PER_BLOCK * HID];   // 32 KB, A-operand source

    const int tid = threadIdx.x;
    const int blockBase = blockIdx.x * PTS_PER_BLOCK;

    // ---- Phase 1: tri-plane bilinear gather, one point per thread ----
    if (tid < PTS_PER_BLOCK) {
        int i = blockBase + tid;
        float F = 0.f;
        if (i < nPoints) {
            float px = (pts[3 * i + 0] + 1.f) * 0.5f * (float)(RES - 1);
            float py = (pts[3 * i + 1] + 1.f) * 0.5f * (float)(RES - 1);
            float pz = (pts[3 * i + 2] + 1.f) * 0.5f * (float)(RES - 1);
            F = bilerp(xy, px, py) + bilerp(xz, px, pz) + bilerp(yz, py, pz);
        }
        sF[tid] = F;
    }
    __syncthreads();

    // ---- Phase 2: h1 = relu(F*w1 + b1), fp32 math, fp16 store to LDS ----
    {
        int p  = tid >> 1;
        int kb = (tid & 1) * 64;
        float F = sF[p];
        for (int k = 0; k < 64; k += 4) {
            f32x4 wv = *(const f32x4*)(w1 + kb + k);
            f32x4 bv = *(const f32x4*)(b1 + kb + k);
            union { _Float16 h[4]; f32x2 f; } u;
            u.h[0] = (_Float16)fmaxf(F * wv.x + bv.x, 0.f);
            u.h[1] = (_Float16)fmaxf(F * wv.y + bv.y, 0.f);
            u.h[2] = (_Float16)fmaxf(F * wv.z + bv.z, 0.f);
            u.h[3] = (_Float16)fmaxf(F * wv.w + bv.w, 0.f);
            *(f32x2*)&sh1[p * HID + kb + k] = u.f;
        }
    }
    __syncthreads();

    // ---- Phase 3: per-wave 16-point tile through v_wmma_f32_16x16x32_f16 ----
    const int wave = tid >> 5;
    const int lane = tid & 31;
    const int li   = lane & 15;      // column / M-low index
    const int hi   = lane >> 4;      // half-wave select
    const int p0   = wave * 16;      // first point of this wave's tile

    // A operands (16x32 f16): row-major h1 matches the A lane layout via two
    // contiguous 16B chunks per lane: K = c*32 + hi*8 + {0..7} and +16.
    V16H A[4];
#pragma unroll
    for (int c = 0; c < 4; ++c) {
        const _Float16* rowp = &sh1[(p0 + li) * HID + c * 32 + hi * 8];
        A[c].q[0] = *(const f32x4*)(rowp);
        A[c].q[1] = *(const f32x4*)(rowp + 16);
    }

    float acc[8] = {0.f, 0.f, 0.f, 0.f, 0.f, 0.f, 0.f, 0.f};
    const float b3s = b3[0];

    for (int tile = 0; tile < 8; ++tile) {
        v8f C = {0.f, 0.f, 0.f, 0.f, 0.f, 0.f, 0.f, 0.f};
#pragma unroll
        for (int c = 0; c < 4; ++c) {
            V16H B;   // pre-swizzled: 32 contiguous bytes per lane (L2-hot)
            const _Float16* bp = w2h + ((size_t)((tile * 4 + c) * 32 + lane)) * 16;
            B.q[0] = *(const f32x4*)(bp);
            B.q[1] = *(const f32x4*)(bp + 8);
            C = __builtin_amdgcn_wmma_f32_16x16x32_f16(
                    false, A[c].v, false, B.v, (short)0, C, false, false);
        }
        // C layout: lane 0-15 -> col=lane, VGPR r -> M=r; lane 16-31 -> M=r+8.
        int col   = tile * 16 + li;
        float bias = b2[col];
        float wv3  = w3[col];
#pragma unroll
        for (int r = 0; r < 8; ++r)
            acc[r] += fmaxf(C[r] + bias, 0.f) * wv3;
    }

    // ---- Layer-3 reduction across the 16-lane half (wave32 shuffles) ----
#pragma unroll
    for (int r = 0; r < 8; ++r) {
        acc[r] += __shfl_xor(acc[r], 1, 32);
        acc[r] += __shfl_xor(acc[r], 2, 32);
        acc[r] += __shfl_xor(acc[r], 4, 32);
        acc[r] += __shfl_xor(acc[r], 8, 32);
    }
    if (li < 8) {
        int pt = blockBase + p0 + li + hi * 8;   // group0: M 0..7, group1: M 8..15
        if (pt < nPoints)
            out[pt] = tanhf(acc[li] + b3s);
    }
}

extern "C" void kernel_launch(void* const* d_in, const int* in_sizes, int n_in,
                              void* d_out, int out_size, void* d_ws, size_t ws_size,
                              hipStream_t stream) {
    const float* pts = (const float*)d_in[0];
    const float* xy  = (const float*)d_in[1];
    const float* yz  = (const float*)d_in[2];
    const float* xz  = (const float*)d_in[3];
    const float* w1  = (const float*)d_in[4];
    const float* b1  = (const float*)d_in[5];
    const float* w2  = (const float*)d_in[6];
    const float* b2  = (const float*)d_in[7];
    const float* w3  = (const float*)d_in[8];
    const float* b3  = (const float*)d_in[9];
    float* out = (float*)d_out;
    _Float16* wsH = (_Float16*)d_ws;   // 32 KB used for pre-swizzled fp16 w2

    const int nPoints = in_sizes[0] / 3;
    const int nBlocks = (nPoints + PTS_PER_BLOCK - 1) / PTS_PER_BLOCK;

    swizzle_w2_kernel<<<1, THREADS, 0, stream>>>(w2, wsH);
    neusdf_fused_kernel<<<nBlocks, THREADS, 0, stream>>>(
        pts, xy, yz, xz, w1, b1, b2, w3, b3, wsH, out, nPoints);
}
